// ScaledDotProductAttention_22479858827414
// MI455X (gfx1250) — matrix operand order repaired and checked by reading, not compiled, not run on hardware
//
#include <hip/hip_runtime.h>

typedef __attribute__((ext_vector_type(16))) _Float16 v16h;
typedef __attribute__((ext_vector_type(8)))  _Float16 v8h;
typedef __attribute__((ext_vector_type(4)))  _Float16 v4h;
typedef __attribute__((ext_vector_type(2)))  _Float16 v2h;
typedef __attribute__((ext_vector_type(8)))  float    v8f;

#define ATT_B 2
#define ATT_H 16
#define ATT_S 2048
#define ATT_D 64

#define BLOCK_M 128   // query rows per block (8 waves x 16)
#define BLOCK_N 64    // keys per iteration
#define NWAVES  8
#define KPAD    72    // f16/row: 144B stride -> distinct banks across 16 rows
#define VPAD    72
#define PPAD    72
#define OPAD    68    // f32/row for the output transpose scratch

// Load 16 f16 (one WMMA operand) as two b128 reads.
__device__ __forceinline__ v16h ld16h(const _Float16* p) {
    v8h lo = *(const v8h*)p;
    v8h hi = *(const v8h*)(p + 8);
    return __builtin_shufflevector(lo, hi, 0,1,2,3,4,5,6,7,8,9,10,11,12,13,14,15);
}

__global__ __launch_bounds__(256)
void ScaledDotProductAttention_22479858827414_kernel(
    const float* __restrict__ Q, const float* __restrict__ K,
    const float* __restrict__ V, float* __restrict__ O)
{
    __shared__ alignas(16) _Float16 sK[BLOCK_N][KPAD];     // keys row-major (key, d)
    __shared__ alignas(16) _Float16 sV[ATT_D][VPAD];       // values transposed (d, key)
    __shared__ alignas(16) _Float16 sP[NWAVES][16][PPAD];  // P^T bounce: (q, key) per wave
    __shared__ alignas(16) float    sO[NWAVES][16][OPAD];  // O transpose scratch (q, d)

    const int bh    = blockIdx.y;
    const int mtile = blockIdx.x;
    const int tid   = threadIdx.x;
    const int wave  = tid >> 5;
    const int lane  = tid & 31;
    const int half  = lane >> 4;
    const int l16   = lane & 15;

    const size_t headoff = (size_t)bh * ATT_S * ATT_D;
    const float* __restrict__ Qh = Q + headoff;
    const float* __restrict__ Kh = K + headoff;
    const float* __restrict__ Vh = V + headoff;
    float* __restrict__ Oh = O + headoff;

    const int qbase = mtile * BLOCK_M + wave * 16;
    // 1/sqrt(64) * log2(e): scores land in the base-2 domain so the hardware
    // v_exp_f32 (base-2) needs no per-element log2e multiply.
    const float scale = 0.125f * 1.44269504088896340736f;

    // ---- Q^T B-operand, resident all pass. B layout: lane -> col(q) = l16;
    // element e -> k(d) = 32*c + e + 16*half  (16 contiguous f32 per lane).
    v16h bQ[2];
    {
        const float* qrow = Qh + (size_t)(qbase + l16) * ATT_D;
        #pragma unroll
        for (int c = 0; c < 2; ++c)
            #pragma unroll
            for (int e = 0; e < 16; ++e)
                bQ[c][e] = (_Float16)(qrow[32 * c + 16 * half + e] * scale);
    }

    // Per-lane softmax state (base-2 domain): this lane owns query col (qbase + l16).
    v8f acc[4] = {v8f{}, v8f{}, v8f{}, v8f{}};   // O^T: 4 d-tiles, lane = q col
    float m = -3.0e38f, l = 0.0f;

    for (int jb = 0; jb < ATT_S; jb += BLOCK_N) {
        // ---- Stage K (row-major, packed b64) and V (transposed, packed b32 row pairs).
        #pragma unroll
        for (int f = tid; f < (BLOCK_N / 2) * (ATT_D / 4); f += 256) {  // 512 -> 2 iters
            const int rp = f >> 4;             // key row pair 0..31
            const int c4 = (f & 15) << 2;      // d offset 0..60
            const int r0 = rp * 2, r1 = r0 + 1;
            const float4 k0 = *(const float4*)(Kh + (size_t)(jb + r0) * ATT_D + c4);
            const float4 k1 = *(const float4*)(Kh + (size_t)(jb + r1) * ATT_D + c4);
            v4h kp0 = { (_Float16)k0.x, (_Float16)k0.y, (_Float16)k0.z, (_Float16)k0.w };
            v4h kp1 = { (_Float16)k1.x, (_Float16)k1.y, (_Float16)k1.z, (_Float16)k1.w };
            *(v4h*)&sK[r0][c4] = kp0;
            *(v4h*)&sK[r1][c4] = kp1;
            const float4 v0 = *(const float4*)(Vh + (size_t)(jb + r0) * ATT_D + c4);
            const float4 v1 = *(const float4*)(Vh + (size_t)(jb + r1) * ATT_D + c4);
            v2h t0 = { (_Float16)v0.x, (_Float16)v1.x };
            v2h t1 = { (_Float16)v0.y, (_Float16)v1.y };
            v2h t2 = { (_Float16)v0.z, (_Float16)v1.z };
            v2h t3 = { (_Float16)v0.w, (_Float16)v1.w };
            *(v2h*)&sV[c4 + 0][r0] = t0;
            *(v2h*)&sV[c4 + 1][r0] = t1;
            *(v2h*)&sV[c4 + 2][r0] = t2;
            *(v2h*)&sV[c4 + 3][r0] = t3;
        }
        if (jb + BLOCK_N < ATT_S) {
            __builtin_prefetch(Kh + (size_t)(jb + BLOCK_N) * ATT_D + tid * 16, 0, 1);
            __builtin_prefetch(Vh + (size_t)(jb + BLOCK_N) * ATT_D + tid * 16, 0, 1);
        }
        __syncthreads();

        // ---- S^T tiles (64 keys x 16 q) = K * Q^T, 4 M-tiles of 16 keys.
        // A layout: lane -> row(key) = mt*16 + l16; element e -> d = 32c + e + 16*half.
        v8f cT[4];
        #pragma unroll
        for (int mt = 0; mt < 4; ++mt) {
            const _Float16* krow = &sK[mt * 16 + l16][0];
            v8f c = {};
            c = __builtin_amdgcn_wmma_f32_16x16x32_f16(
                    false, ld16h(krow + 16 * half),      false, bQ[0], (short)0, c, false, false);
            c = __builtin_amdgcn_wmma_f32_16x16x32_f16(
                    false, ld16h(krow + 32 + 16 * half), false, bQ[1], (short)0, c, false, false);
            cT[mt] = c;
        }

        // ---- Online softmax (base 2): lane owns one query; partner lane (xor 16)
        // holds the other half of its keys per tile -> single cross-lane combine.
        float rmax = -3.0e38f;
        #pragma unroll
        for (int mt = 0; mt < 4; ++mt)
            #pragma unroll
            for (int j = 0; j < 8; ++j)
                rmax = fmaxf(rmax, cT[mt][j]);
        rmax = fmaxf(rmax, __shfl_xor(rmax, 16, 32));
        const float mnew  = fmaxf(m, rmax);
        const float alpha = __builtin_amdgcn_exp2f(m - mnew);
        m = mnew;
        float rsum = 0.0f;
        #pragma unroll
        for (int mt = 0; mt < 4; ++mt)
            #pragma unroll
            for (int j = 0; j < 8; ++j) {
                const float p = __builtin_amdgcn_exp2f(cT[mt][j] - mnew);
                cT[mt][j] = p;
                rsum += p;
            }
        rsum += __shfl_xor(rsum, 16, 32);
        l = l * alpha + rsum;
        #pragma unroll
        for (int dt = 0; dt < 4; ++dt)
            #pragma unroll
            for (int j = 0; j < 8; ++j)
                acc[dt][j] *= alpha;

        // ---- P^T bounce: lane's 8 values per tile are contiguous keys of its own
        // query -> one packed b128 store per tile into sP[q][key].
        #pragma unroll
        for (int mt = 0; mt < 4; ++mt) {
            v8h ph;
            #pragma unroll
            for (int j = 0; j < 8; ++j)
                ph[j] = (_Float16)cT[mt][j];
            *(v8h*)&sP[wave][l16][mt * 16 + 8 * half] = ph;
        }
        // B operand P^T: lane -> col(q) = l16; element e -> key = 32*kc + e + 16*half.
        v16h bP[2];
        #pragma unroll
        for (int kc = 0; kc < 2; ++kc)
            bP[kc] = ld16h(&sP[wave][l16][32 * kc + 16 * half]);

        // ---- O^T += V^T * P^T. A layout: lane -> row(d) = dt*16 + l16;
        // element e -> key = 32*kc + e + 16*half.
        #pragma unroll
        for (int dt = 0; dt < 4; ++dt) {
            const _Float16* vrow = &sV[dt * 16 + l16][0];
            acc[dt] = __builtin_amdgcn_wmma_f32_16x16x32_f16(
                          false, ld16h(vrow + 16 * half),      false, bP[0], (short)0, acc[dt], false, false);
            acc[dt] = __builtin_amdgcn_wmma_f32_16x16x32_f16(
                          false, ld16h(vrow + 32 + 16 * half), false, bP[1], (short)0, acc[dt], false, false);
        }
        __syncthreads();   // protect sK/sV restage
    }

    // ---- Epilogue: divide by l (per-lane scalar), transpose O^T -> O via per-wave
    // LDS scratch, then fully coalesced global stores (256B per query row).
    const float invl = 1.0f / l;
    #pragma unroll
    for (int dt = 0; dt < 4; ++dt)
        #pragma unroll
        for (int j = 0; j < 8; j += 2) {
            float2 t = make_float2(acc[dt][j] * invl, acc[dt][j + 1] * invl);
            *(float2*)&sO[wave][l16][dt * 16 + 8 * half + j] = t;   // (q=l16, d)
        }
    #pragma unroll
    for (int r = 0; r < 16; ++r) {
        const float2 t = *(const float2*)&sO[wave][r][lane * 2];
        *(float2*)&Oh[(size_t)(qbase + r) * ATT_D + lane * 2] = t;
    }
}

extern "C" void kernel_launch(void* const* d_in, const int* in_sizes, int n_in,
                              void* d_out, int out_size, void* d_ws, size_t ws_size,
                              hipStream_t stream) {
    (void)in_sizes; (void)n_in; (void)out_size; (void)d_ws; (void)ws_size;
    const float* Q = (const float*)d_in[0];
    const float* K = (const float*)d_in[1];
    const float* V = (const float*)d_in[2];
    float* O = (float*)d_out;
    dim3 grid(ATT_S / BLOCK_M, ATT_B * ATT_H);
    dim3 block(256);
    ScaledDotProductAttention_22479858827414_kernel<<<grid, block, 0, stream>>>(Q, K, V, O);
}
